// FuncGNN_46763603919180
// MI455X (gfx1250) — matrix operand order repaired and checked by reading, not compiled, not run on hardware
//
#include <hip/hip_runtime.h>
#include <hip/hip_bf16.h>

// ---------------------------------------------------------------------------
// EGNN forward for MI455X (gfx1250, wave32, WMMA).
// All GEMMs: v_wmma_f32_16x16x32_f16, f32 accumulation, 32-row M tiles per
// wave (two A fragments share each B fragment; B loads software-pipelined).
// ---------------------------------------------------------------------------

typedef __attribute__((ext_vector_type(16))) _Float16 v16h;
typedef __attribute__((ext_vector_type(8)))  _Float16 v8h;
typedef __attribute__((ext_vector_type(8)))  float    v8f;

#define NN  16384
#define EE  262144
#define HIDN 128

// ---------------- WMMA fragment helpers (layouts per CDNA5 ISA) --

__device__ inline int laneId() { return threadIdx.x & 31; }

// A-matrix 16x32 f16 fragment from LDS tile stored row-major [16][kpad].
__device__ inline v16h a_frag(const _Float16* a, int kpad, int k0) {
  const int lane = laneId();
  const _Float16* p = a + (lane & 15) * kpad + k0 + ((lane >> 4) << 3);
  v8h lo = *(const v8h*)p;        // ds_load_b128
  v8h hi = *(const v8h*)(p + 16); // ds_load_b128
  v16h r;
#pragma unroll
  for (int i = 0; i < 8; ++i) { r[i] = lo[i]; r[i + 8] = hi[i]; }
  return r;
}

__device__ inline void zero_acc8(v8f acc[8]) {
#pragma unroll
  for (int nt = 0; nt < 8; ++nt)
#pragma unroll
    for (int q = 0; q < 8; ++q) acc[nt][q] = 0.f;
}

// D[32,128] += A[32,kchunks*32] * B.  B pre-packed in fragment order:
// packed[((kc*8+nt)*32 + lane)*16 + i] = W[kc*32+(lane>>4)*8+(i&7)+(i>>3)*16][nt*16+(lane&15)]
// Two 16-row A tiles share each B fragment; next B fragment is prefetched
// one step ahead so global_load_b128 overlaps the WMMA pair.
__device__ inline void gemm_acc8x2(const _Float16* aLds, int kpad, int kchunks,
                                   const _Float16* bp, v8f acc0[8], v8f acc1[8]) {
  const int lane = laneId();
  const v16h* bv = (const v16h*)bp;
  const int total = kchunks * 8;
  v16h b = bv[lane];                       // fragment 0
  for (int kc = 0; kc < kchunks; ++kc) {
    v16h a0 = a_frag(aLds, kpad, kc * 32);
    v16h a1 = a_frag(aLds + 16 * kpad, kpad, kc * 32);
#pragma unroll
    for (int nt = 0; nt < 8; ++nt) {
      int idx = kc * 8 + nt;
      v16h bcur = b;
      if (idx + 1 < total) b = bv[(size_t)(idx + 1) * 32 + lane];  // prefetch
      acc0[nt] = __builtin_amdgcn_wmma_f32_16x16x32_f16(
          false, a0, false, bcur, (short)0, acc0[nt], false, false);
      acc1[nt] = __builtin_amdgcn_wmma_f32_16x16x32_f16(
          false, a1, false, bcur, (short)0, acc1[nt], false, false);
    }
  }
}

// C/D layout: element rr of acc[nt] in lane = row (lane>>4)*8+rr, col nt*16+(lane&15)
__device__ inline void store_act_f16_2(v8f acc0[8], v8f acc1[8],
                                       const float* bias, int act,
                                       _Float16* dst /*32x128 row-major LDS*/) {
  const int lane = laneId(), n0 = lane & 15, mb = (lane >> 4) << 3;
#pragma unroll
  for (int nt = 0; nt < 8; ++nt) {
    float bb = bias[nt * 16 + n0];
#pragma unroll
    for (int rr = 0; rr < 8; ++rr) {
      float v0 = acc0[nt][rr] + bb;
      float v1 = acc1[nt][rr] + bb;
      if (act == 1) { v0 = v0 / (1.f + __expf(-v0)); v1 = v1 / (1.f + __expf(-v1)); }
      else if (act == 2) { v0 = v0 > 0.f ? v0 : 0.f; v1 = v1 > 0.f ? v1 : 0.f; }
      dst[(mb + rr) * HIDN + nt * 16 + n0] = (_Float16)v0;
      dst[(16 + mb + rr) * HIDN + nt * 16 + n0] = (_Float16)v1;
    }
  }
}

// ---------------- utility kernels ------------------------------------------

__global__ void zero_f32_kernel(float* d, int n) {
  int i = blockIdx.x * blockDim.x + threadIdx.x;
  if (i < n) d[i] = 0.f;
}
__global__ void copy_f32_kernel(const float* s, float* d, int n) {
  int i = blockIdx.x * blockDim.x + threadIdx.x;
  if (i < n) d[i] = s[i];
}
__global__ void f32_to_f16_kernel(const float* s, _Float16* d, int n) {
  int i = blockIdx.x * blockDim.x + threadIdx.x;
  if (i < n) d[i] = (_Float16)s[i];
}

// Pack row-major f32 weight [K][N] -> f16 WMMA B-fragment order, pad [Kp][Np].
__global__ void pack_w_kernel(const float* __restrict__ w, _Float16* __restrict__ dst,
                              int K, int N, int Kp, int Np) {
  int NT = Np >> 4;
  int total = Kp * Np;
  for (int idx = blockIdx.x * blockDim.x + threadIdx.x; idx < total;
       idx += gridDim.x * blockDim.x) {
    int i = idx & 15;
    int lane = (idx >> 4) & 31;
    int t = idx >> 9;            // kc*NT + nt
    int nt = t % NT, kc = t / NT;
    int n = nt * 16 + (lane & 15);
    int k = kc * 32 + ((lane >> 4) << 3) + (i & 7) + ((i >> 3) << 4);
    dst[idx] = (k < K && n < N) ? (_Float16)w[(size_t)k * N + n] : (_Float16)0.f;
  }
}

// ---------------- edge model kernel (E_GCL edge + coord half) ---------------
// One wave (one workgroup) = 32 edges.  e_in = [h[row]|h[col]|radial|ea|pad].
// e1->SiLU->e2->SiLU => m ; f32 atomics into magg ; optionally c1->SiLU->c2
// => cm, atomics into xagg/cnt.
__global__ __launch_bounds__(32) void edge_gcl_kernel(
    const _Float16* __restrict__ hf16, const float* __restrict__ x,
    const int* __restrict__ ei, const float* __restrict__ ea,
    const _Float16* __restrict__ w1p, const float* __restrict__ b1,
    const _Float16* __restrict__ w2p, const float* __restrict__ b2,
    const _Float16* __restrict__ wc1p, const float* __restrict__ bc1,
    const float* __restrict__ wc2,
    float* __restrict__ magg, float* __restrict__ xagg, float* __restrict__ cnt,
    int upd) {
  __shared__ _Float16 Ast[32 * 288];   // gathered e_in
  __shared__ _Float16 Mst[32 * 128];   // m (e2 out)
  __shared__ _Float16 Qst[32 * 128];   // m1 (e1 out), later q (c1 out)
  __shared__ float    Cd[32][4];
  __shared__ int      Row[32];
  __shared__ float    Wc2s[128];

  const int lane = laneId();
  const int ge = blockIdx.x * 32 + lane;          // one edge per lane
  const int r = ei[ge], c = ei[EE + ge];
  Row[lane] = r;
  {
    const v8h* hr = (const v8h*)(hf16 + (size_t)r * HIDN);
    const v8h* hc = (const v8h*)(hf16 + (size_t)c * HIDN);
    v8h* dst = (v8h*)&Ast[lane * 288];
#pragma unroll
    for (int i = 0; i < 16; ++i) dst[i] = hr[i];
#pragma unroll
    for (int i = 0; i < 16; ++i) dst[16 + i] = hc[i];
  }
  {
    float d0 = x[r * 3 + 0] - x[c * 3 + 0];
    float d1 = x[r * 3 + 1] - x[c * 3 + 1];
    float d2 = x[r * 3 + 2] - x[c * 3 + 2];
    Cd[lane][0] = d0; Cd[lane][1] = d1; Cd[lane][2] = d2;
    _Float16* t = &Ast[lane * 288];
    t[256] = (_Float16)(d0 * d0 + d1 * d1 + d2 * d2);
#pragma unroll
    for (int i = 0; i < 16; ++i) t[257 + i] = (_Float16)ea[(size_t)ge * 16 + i];
#pragma unroll
    for (int i = 273; i < 288; ++i) t[i] = (_Float16)0.f;
  }
#pragma unroll
  for (int i = 0; i < 4; ++i) Wc2s[lane * 4 + i] = wc2[lane * 4 + i];
  __syncthreads();   // single-wave WG: barrier -> S_NOP, keeps dscnt fence

  const int n0 = lane & 15, mb = (lane >> 4) << 3;
  v8f acc0[8], acc1[8];

  zero_acc8(acc0); zero_acc8(acc1);
  gemm_acc8x2(&Ast[0], 288, 9, w1p, acc0, acc1);     // e1: 144 WMMAs
  store_act_f16_2(acc0, acc1, b1, 1, &Qst[0]);
  __syncthreads();

  zero_acc8(acc0); zero_acc8(acc1);
  gemm_acc8x2(&Qst[0], 128, 4, w2p, acc0, acc1);     // e2: 64 WMMAs
#pragma unroll
  for (int nt = 0; nt < 8; ++nt) {
    float bb = b2[nt * 16 + n0];
#pragma unroll
    for (int rr = 0; rr < 8; ++rr) {
      float v0 = acc0[nt][rr] + bb;  v0 = v0 / (1.f + __expf(-v0));
      float v1 = acc1[nt][rr] + bb;  v1 = v1 / (1.f + __expf(-v1));
      int col = nt * 16 + n0;
      Mst[(mb + rr) * HIDN + col] = (_Float16)v0;
      Mst[(16 + mb + rr) * HIDN + col] = (_Float16)v1;
      atomicAdd(&magg[(size_t)Row[mb + rr] * HIDN + col], v0);
      atomicAdd(&magg[(size_t)Row[16 + mb + rr] * HIDN + col], v1);
    }
  }
  __syncthreads();

  if (upd) {  // coord model
    zero_acc8(acc0); zero_acc8(acc1);
    gemm_acc8x2(&Mst[0], 128, 4, wc1p, acc0, acc1);  // c1: 64 WMMAs
    store_act_f16_2(acc0, acc1, bc1, 1, &Qst[0]);
    __syncthreads();
    // c2: one lane = one edge, full 128-dot
    float s = 0.f;
    const _Float16* q = &Qst[lane * 128];
#pragma unroll
    for (int j = 0; j < 128; ++j) s += (float)q[j] * Wc2s[j];
    atomicAdd(&xagg[r * 4 + 0], Cd[lane][0] * s);
    atomicAdd(&xagg[r * 4 + 1], Cd[lane][1] * s);
    atomicAdd(&xagg[r * 4 + 2], Cd[lane][2] * s);
    atomicAdd(&cnt[r], 1.f);
  }
}

// ---------------- node model kernel ----------------------------------------
// One wave (one workgroup) = 32 nodes.  A = [h | magg | h0]; n1->SiLU->n2;
// h += out (residual); x += xagg/max(cnt,1); reset aggregation buffers.
__global__ __launch_bounds__(32) void node_gcl_kernel(
    float* __restrict__ hcur, _Float16* __restrict__ hf16,
    const _Float16* __restrict__ h0f16, int nf0,
    float* __restrict__ magg, float* __restrict__ xagg, float* __restrict__ cnt,
    float* __restrict__ x,
    const _Float16* __restrict__ w1p, const float* __restrict__ b1,
    const _Float16* __restrict__ w2p, const float* __restrict__ b2,
    int kpad, int updx) {
  __shared__ _Float16 Ast[32 * 384];
  __shared__ _Float16 Tst[32 * 128];
  const int lane = laneId();
  const int m0 = blockIdx.x * 32;
  const int m = m0 + lane;                       // one node per lane
  _Float16* arow = &Ast[lane * kpad];
  {
    const v8h* hp = (const v8h*)(hf16 + (size_t)m * HIDN);
    v8h* dst = (v8h*)arow;
#pragma unroll
    for (int i = 0; i < 16; ++i) dst[i] = hp[i];
  }
  {
    const float4* mg = (const float4*)(magg + (size_t)m * HIDN);
    _Float16* d1 = arow + 128;
#pragma unroll
    for (int i = 0; i < 32; ++i) {
      float4 f = mg[i];
      d1[i * 4 + 0] = (_Float16)f.x; d1[i * 4 + 1] = (_Float16)f.y;
      d1[i * 4 + 2] = (_Float16)f.z; d1[i * 4 + 3] = (_Float16)f.w;
    }
  }
  if (nf0 > 0) {
    const _Float16* h0 = h0f16 + (size_t)m * nf0;
    _Float16* d2 = arow + 256;
    for (int i = 0; i < nf0; ++i) d2[i] = h0[i];
  }
  __syncthreads();

  const int n0 = lane & 15, mb = (lane >> 4) << 3;
  v8f acc0[8], acc1[8];
  zero_acc8(acc0); zero_acc8(acc1);
  gemm_acc8x2(&Ast[0], kpad, kpad >> 5, w1p, acc0, acc1);   // n1
  store_act_f16_2(acc0, acc1, b1, 1, &Tst[0]);
  __syncthreads();

  zero_acc8(acc0); zero_acc8(acc1);
  gemm_acc8x2(&Tst[0], 128, 4, w2p, acc0, acc1);            // n2
#pragma unroll
  for (int nt = 0; nt < 8; ++nt) {
    float bb = b2[nt * 16 + n0];
#pragma unroll
    for (int rr = 0; rr < 8; ++rr) {
      int col = nt * 16 + n0;
      size_t o0 = (size_t)(m0 + mb + rr) * HIDN + col;
      size_t o1 = (size_t)(m0 + 16 + mb + rr) * HIDN + col;
      float v0 = hcur[o0] + acc0[nt][rr] + bb;      // recurrent residual
      float v1 = hcur[o1] + acc1[nt][rr] + bb;
      hcur[o0] = v0; hf16[o0] = (_Float16)v0;
      hcur[o1] = v1; hf16[o1] = (_Float16)v1;
    }
  }
  // housekeeping: reset aggregation buffers, apply coord update
  {
    float4* mg = (float4*)(magg + (size_t)m * HIDN);
    float4 z4 = {0.f, 0.f, 0.f, 0.f};
#pragma unroll
    for (int i = 0; i < 32; ++i) mg[i] = z4;
  }
  if (updx) {
    float cc = cnt[m]; cc = cc > 1.f ? cc : 1.f;
    x[m * 3 + 0] += xagg[m * 4 + 0] / cc;
    x[m * 3 + 1] += xagg[m * 4 + 1] / cc;
    x[m * 3 + 2] += xagg[m * 4 + 2] / cc;
  }
  cnt[m] = 0.f;
  xagg[m * 4 + 0] = 0.f; xagg[m * 4 + 1] = 0.f;
  xagg[m * 4 + 2] = 0.f; xagg[m * 4 + 3] = 0.f;
}

// ---------------- block embedding (plain Linear) ----------------------------
__global__ __launch_bounds__(32) void embed_kernel(
    const _Float16* __restrict__ h0f16, int nf0,
    const _Float16* __restrict__ wp, const float* __restrict__ bias,
    float* __restrict__ hcur, _Float16* __restrict__ hf16) {
  __shared__ _Float16 Ast[32 * 128];
  const int lane = laneId();
  const int m0 = blockIdx.x * 32;
  const int m = m0 + lane;
  {
    const _Float16* s = h0f16 + (size_t)m * nf0;
    _Float16* d = &Ast[lane * nf0];
    for (int i = 0; i < nf0; ++i) d[i] = s[i];
  }
  __syncthreads();
  v8f acc0[8], acc1[8];
  zero_acc8(acc0); zero_acc8(acc1);
  gemm_acc8x2(&Ast[0], nf0, nf0 >> 5, wp, acc0, acc1);
  const int n0 = lane & 15, mb = (lane >> 4) << 3;
#pragma unroll
  for (int nt = 0; nt < 8; ++nt) {
    float bb = bias[nt * 16 + n0];
#pragma unroll
    for (int rr = 0; rr < 8; ++rr) {
      int col = nt * 16 + n0;
      size_t o0 = (size_t)(m0 + mb + rr) * HIDN + col;
      size_t o1 = (size_t)(m0 + 16 + mb + rr) * HIDN + col;
      float v0 = acc0[nt][rr] + bb;
      float v1 = acc1[nt][rr] + bb;
      hcur[o0] = v0; hf16[o0] = (_Float16)v0;
      hcur[o1] = v1; hf16[o1] = (_Float16)v1;
    }
  }
}

// ---------------- global mean pool (contiguous 1024-node segments) ----------
__global__ void pool_kernel(const float* __restrict__ hcur,
                            float* __restrict__ pooled,
                            _Float16* __restrict__ pooledf16) {
  int b = blockIdx.x, f = threadIdx.x;   // 16 blocks x 128 threads
  float s = 0.f;
  const float* base = hcur + (size_t)b * 1024 * HIDN + f;
  for (int i = 0; i < 1024; ++i) s += base[(size_t)i * HIDN];
  s *= (1.0f / 1024.0f);
  pooled[b * HIDN + f] = s;
  pooledf16[b * HIDN + f] = (_Float16)s;
}

// ---------------- task-MLP head (128 rows, 192->128->128->128->3) -----------
__global__ __launch_bounds__(64) void head_kernel(
    const _Float16* __restrict__ pooledf16, const _Float16* __restrict__ tembf16,
    const int* __restrict__ ti,
    const _Float16* w0p, const float* b0,
    const _Float16* w1p, const float* b1,
    const _Float16* w2p, const float* b2,
    const _Float16* w3p, const float* b3,
    float* __restrict__ out) {
  __shared__ _Float16 Ast[2][32 * 192];
  __shared__ _Float16 Bst[2][32 * 128];
  const int w = threadIdx.x >> 5, lane = laneId();
  const int r0 = (blockIdx.x * 2 + w) * 32;          // 32 rows per wave
  const int gr = r0 + lane;
  {
    int prot = ti[gr * 2];
    int task = ti[gr * 2 + 1];
    const v8h* sp = (const v8h*)(pooledf16 + (size_t)prot * 128);
    const v8h* st = (const v8h*)(tembf16 + (size_t)task * 64);
    v8h* d = (v8h*)&Ast[w][lane * 192];
#pragma unroll
    for (int i = 0; i < 16; ++i) d[i] = sp[i];
#pragma unroll
    for (int i = 0; i < 8; ++i) d[16 + i] = st[i];
  }
  __syncthreads();
  const int n0 = lane & 15, mb = (lane >> 4) << 3;
  v8f acc0[8], acc1[8];
  zero_acc8(acc0); zero_acc8(acc1);
  gemm_acc8x2(&Ast[w][0], 192, 6, w0p, acc0, acc1);
  store_act_f16_2(acc0, acc1, b0, 2, &Bst[w][0]);
  __syncthreads();
  zero_acc8(acc0); zero_acc8(acc1);
  gemm_acc8x2(&Bst[w][0], 128, 4, w1p, acc0, acc1);
  store_act_f16_2(acc0, acc1, b1, 2, &Ast[w][0]);    // reuse region, kpad=128
  __syncthreads();
  zero_acc8(acc0); zero_acc8(acc1);
  gemm_acc8x2(&Ast[w][0], 128, 4, w2p, acc0, acc1);
  store_act_f16_2(acc0, acc1, b2, 2, &Bst[w][0]);
  __syncthreads();
  // final layer: N padded 3->16 (single n-tile), two 16-row tiles
  v8f a30, a31;
#pragma unroll
  for (int q = 0; q < 8; ++q) { a30[q] = 0.f; a31[q] = 0.f; }
  for (int kc = 0; kc < 4; ++kc) {
    v16h a0 = a_frag(&Bst[w][0], 128, kc * 32);
    v16h a1 = a_frag(&Bst[w][16 * 128], 128, kc * 32);
    v16h b = ((const v16h*)w3p)[kc * 32 + lane];
    a30 = __builtin_amdgcn_wmma_f32_16x16x32_f16(false, a0, false, b, (short)0,
                                                 a30, false, false);
    a31 = __builtin_amdgcn_wmma_f32_16x16x32_f16(false, a1, false, b, (short)0,
                                                 a31, false, false);
  }
  if (n0 < 3) {
    float bb = b3[n0];
#pragma unroll
    for (int rr = 0; rr < 8; ++rr) {
      out[(r0 + mb + rr) * 3 + n0] = a30[rr] + bb;
      out[(r0 + 16 + mb + rr) * 3 + n0] = a31[rr] + bb;
    }
  }
}

// ---------------------------------------------------------------------------
// Host driver.
// d_in leaf order assumption (depth-first, dict insertion order):
//  0 h, 1 x, 2 edge_index, 3 edge_attr, 4 batch, 5 tasks_indices,
//  6 batch_size, 7 edge_type, then params:
//  for blk 0..2: emb.w, emb.b, then 4x gcl: e1.w,e1.b,e2.w,e2.b,c1.w,c1.b,
//    c2.w, n1.w,n1.b,n2.w,n2.b
//  pool: e1.w,e1.b,e2.w,e2.b,n1.w,n1.b,n2.w,n2.b
//  mlp: 4x (w,b); task_embed
// ---------------------------------------------------------------------------
extern "C" void kernel_launch(void* const* d_in, const int* in_sizes, int n_in,
                              void* d_out, int out_size, void* d_ws, size_t ws_size,
                              hipStream_t stream) {
  (void)in_sizes; (void)n_in; (void)out_size; (void)ws_size;
  const float* in_h = (const float*)d_in[0];
  const float* in_x = (const float*)d_in[1];
  const int*   ei   = (const int*)d_in[2];
  const float* ea   = (const float*)d_in[3];
  const int*   ti   = (const int*)d_in[5];

  // --- param leaf indices ---
  struct G { int e1w, e1b, e2w, e2b, c1w, c1b, c2w, n1w, n1b, n2w, n2b; };
  int embw[3], embb[3]; G g[3][4];
  int pI = 8;
  for (int b = 0; b < 3; ++b) {
    embw[b] = pI++; embb[b] = pI++;
    for (int gi = 0; gi < 4; ++gi) {
      g[b][gi] = {pI, pI + 1, pI + 2, pI + 3, pI + 4, pI + 5, pI + 6,
                  pI + 7, pI + 8, pI + 9, pI + 10};
      pI += 11;
    }
  }
  int pe1w = pI++, pe1b = pI++, pe2w = pI++, pe2b = pI++;
  int pn1w = pI++, pn1b = pI++, pn2w = pI++, pn2b = pI++;
  int mlpw[4], mlpb[4];
  for (int i = 0; i < 4; ++i) { mlpw[i] = pI++; mlpb[i] = pI++; }
  int teI = pI++;

  // --- workspace bump allocator (order fixed => deterministic) ---
  size_t off = 0;
  auto alloc = [&](size_t bytes) -> char* {
    off = (off + 255) & ~(size_t)255;
    char* p = (char*)d_ws + off;
    off += bytes;
    return p;
  };
  float*    hcur   = (float*)alloc((size_t)NN * HIDN * 4);
  _Float16* hf16   = (_Float16*)alloc((size_t)NN * HIDN * 2);
  _Float16* h0f16  = (_Float16*)alloc((size_t)NN * HIDN * 2);
  float*    magg   = (float*)alloc((size_t)NN * HIDN * 4);
  float*    xagg   = (float*)alloc((size_t)NN * 4 * 4);
  float*    cntb   = (float*)alloc((size_t)NN * 4);
  float*    xcur   = (float*)alloc((size_t)NN * 3 * 4);
  float*    pooled = (float*)alloc(16 * HIDN * 4);
  _Float16* pooledf16 = (_Float16*)alloc(16 * HIDN * 2);
  _Float16* tembf16   = (_Float16*)alloc(100 * 64 * 2);

  auto pack = [&](int leaf, int K, int Nc, int Kp, int Np) -> const _Float16* {
    _Float16* dst = (_Float16*)alloc((size_t)Kp * Np * 2);
    int total = Kp * Np;
    pack_w_kernel<<<(total + 255) / 256, 256, 0, stream>>>(
        (const float*)d_in[leaf], dst, K, Nc, Kp, Np);
    return dst;
  };

  const _Float16 *embP[3], *e1P[3][4], *e2P[3][4], *c1P[3][4], *n1P[3][4], *n2P[3][4];
  for (int b = 0; b < 3; ++b) {
    int nf = (b == 0) ? 32 : 128;
    embP[b] = pack(embw[b], nf, 128, nf, 128);
    for (int gi = 0; gi < 4; ++gi) {
      e1P[b][gi] = pack(g[b][gi].e1w, 273, 128, 288, 128);
      e2P[b][gi] = pack(g[b][gi].e2w, 128, 128, 128, 128);
      c1P[b][gi] = pack(g[b][gi].c1w, 128, 128, 128, 128);
      n1P[b][gi] = pack(g[b][gi].n1w, 256 + nf, 128, 256 + nf, 128);
      n2P[b][gi] = pack(g[b][gi].n2w, 128, 128, 128, 128);
    }
  }
  const _Float16* pe1P = pack(pe1w, 273, 128, 288, 128);
  const _Float16* pe2P = pack(pe2w, 128, 128, 128, 128);
  const _Float16* pn1P = pack(pn1w, 256, 128, 256, 128);
  const _Float16* pn2P = pack(pn2w, 128, 128, 128, 128);
  const _Float16* m0P = pack(mlpw[0], 192, 128, 192, 128);
  const _Float16* m1P = pack(mlpw[1], 128, 128, 128, 128);
  const _Float16* m2P = pack(mlpw[2], 128, 128, 128, 128);
  const _Float16* m3P = pack(mlpw[3], 128, 3, 128, 16);
  f32_to_f16_kernel<<<(100 * 64 + 255) / 256, 256, 0, stream>>>(
      (const float*)d_in[teI], tembf16, 100 * 64);

  // --- init mutable state (fresh every call; graph-replay safe) ---
  zero_f32_kernel<<<(NN * HIDN + 255) / 256, 256, 0, stream>>>(magg, NN * HIDN);
  zero_f32_kernel<<<(NN * 4 + 255) / 256, 256, 0, stream>>>(xagg, NN * 4);
  zero_f32_kernel<<<(NN + 255) / 256, 256, 0, stream>>>(cntb, NN);
  copy_f32_kernel<<<(NN * 3 + 255) / 256, 256, 0, stream>>>(in_x, xcur, NN * 3);
  f32_to_f16_kernel<<<(NN * 32 + 255) / 256, 256, 0, stream>>>(in_h, h0f16, NN * 32);

  // --- 3 EGNN blocks x (embed + 4 GCLs) ---
  int nf0 = 32;
  for (int b = 0; b < 3; ++b) {
    if (b > 0) {
      hipMemcpyAsync(h0f16, hf16, (size_t)NN * HIDN * 2,
                     hipMemcpyDeviceToDevice, stream);
      nf0 = 128;
    }
    embed_kernel<<<NN / 32, 32, 0, stream>>>(h0f16, nf0, embP[b],
                                             (const float*)d_in[embb[b]],
                                             hcur, hf16);
    int kpadN = 256 + nf0;
    for (int gi = 0; gi < 4; ++gi) {
      edge_gcl_kernel<<<EE / 32, 32, 0, stream>>>(
          hf16, xcur, ei, ea,
          e1P[b][gi], (const float*)d_in[g[b][gi].e1b],
          e2P[b][gi], (const float*)d_in[g[b][gi].e2b],
          c1P[b][gi], (const float*)d_in[g[b][gi].c1b],
          (const float*)d_in[g[b][gi].c2w],
          magg, xagg, cntb, 1);
      node_gcl_kernel<<<NN / 32, 32, 0, stream>>>(
          hcur, hf16, h0f16, nf0, magg, xagg, cntb, xcur,
          n1P[b][gi], (const float*)d_in[g[b][gi].n1b],
          n2P[b][gi], (const float*)d_in[g[b][gi].n2b],
          kpadN, 1);
    }
  }

  // --- pooling GCL (no coord update, no node_attr) ---
  edge_gcl_kernel<<<EE / 32, 32, 0, stream>>>(
      hf16, xcur, ei, ea,
      pe1P, (const float*)d_in[pe1b],
      pe2P, (const float*)d_in[pe2b],
      pe2P, (const float*)d_in[pe2b],     // dummies (upd=0, never used)
      (const float*)d_in[pe2b],           // dummy wc2 (128 floats, safe read)
      magg, xagg, cntb, 0);
  node_gcl_kernel<<<NN / 32, 32, 0, stream>>>(
      hcur, hf16, hf16, 0, magg, xagg, cntb, xcur,
      pn1P, (const float*)d_in[pn1b],
      pn2P, (const float*)d_in[pn2b],
      256, 0);

  // --- global mean pool + task head ---
  pool_kernel<<<16, 128, 0, stream>>>(hcur, pooled, pooledf16);
  head_kernel<<<2, 64, 0, stream>>>(
      pooledf16, tembf16, ti,
      m0P, (const float*)d_in[mlpb[0]],
      m1P, (const float*)d_in[mlpb[1]],
      m2P, (const float*)d_in[mlpb[2]],
      m3P, (const float*)d_in[mlpb[3]],
      (float*)d_out);
}
